// HTGNNLayer_22247930593793
// MI455X (gfx1250) — matrix operand order repaired and checked by reading, not compiled, not run on hardware
//
#include <hip/hip_runtime.h>
#include <hip/hip_bf16.h>

// ---------------------------------------------------------------------------
// HTGNN layer for MI455X (gfx1250, wave32).
// Dense GEMMs: v_wmma_f32_16x16x32_f16, 64x64 block tile, 4 WMMA/wave/K-step,
// b128-vectorized global->LDS staging, LDS-staged coalesced b128 epilogue
// stores, global_prefetch of the next K slice.
// Edge softmax-scatter: ordered-uint atomicMax (segment max) + atomic_add_f32.
// ---------------------------------------------------------------------------

typedef __attribute__((ext_vector_type(16))) _Float16 v16h;
typedef __attribute__((ext_vector_type(4)))  _Float16 v4h;
typedef __attribute__((ext_vector_type(8)))  float    v8f;

#define T_FRAMES 4
#define N_NODES  40000
#define N_EDGES  200000
#define N_REL    2
#define N_INP    128
#define N_HID    64

// ---------------- workspace layout (bytes) ----------------
// Region A (81.92MB): intra[R,T,N,64]  -> later q (off 0), k (off+40.96MB) -> later act
// Region B (40.96MB): inter[T,N,64]    -> later v -> later res
// Region C (40.96MB): hproj[T,N,64]    -> later h_att
#define OFF_INTRA   ((size_t)0)
#define OFF_Q       ((size_t)0)
#define OFF_K       ((size_t)40960000)
#define OFF_ACT     ((size_t)0)
#define OFF_INTER   ((size_t)81920000)
#define OFF_V       OFF_INTER
#define OFF_RES     OFF_INTER
#define OFF_HPROJ   ((size_t)122880000)
#define OFF_HATT    OFF_HPROJ
#define OFF_XW      ((size_t)163840000)
#define OFF_OUTACC  ((size_t)184320000)
#define OFF_ALS     ((size_t)204800000)
#define OFF_ALD     ((size_t)205120000)
#define OFF_MENC    ((size_t)205440000)
#define OFF_MF      ((size_t)205760000)
#define OFF_DEN     ((size_t)206080000)
#define OFF_WSUM    ((size_t)206400000)
#define OFF_BETA    ((size_t)206400064)
// total ~206.4 MB

// monotonic float<->uint encoding for atomic segment-max over floats
__device__ __forceinline__ unsigned fenc(float f) {
    unsigned u = __float_as_uint(f);
    return (u & 0x80000000u) ? ~u : (u | 0x80000000u);
}
__device__ __forceinline__ float fdec(unsigned u) {
    return (u & 0x80000000u) ? __uint_as_float(u & 0x7FFFFFFFu)
                             : __uint_as_float(~u);
}

// ---------------------------------------------------------------------------
// WMMA GEMM: C[M x N] = A[M x K] * B (+ epilogue)
//   transB==0 : B stored [K][N]            (GAT x@W)
//   transB==1 : B stored [N][K] (A @ B^T)  (all torch Linear)
// epilogue mode: 0 none | 1 +bias | 2 +bias+PE(t=row/nodesPerFrame) | 3 relu(+bias)
// grid = (M/64, N/64), block = 128 (4 waves).
// Each wave: 16 M-rows x 64 N-cols = 4 accumulators; A frag reused 4x.
// ---------------------------------------------------------------------------
__global__ __launch_bounds__(128)
void wmma_gemm_f32(const float* __restrict__ A, const float* __restrict__ B,
                   float* __restrict__ C, const float* __restrict__ bias,
                   int M, int N, int Kdim, int transB, int mode, int nodesPerFrame)
{
    __shared__ __align__(16) _Float16 lA[64 * 32];   // [row][k]
    __shared__ __align__(16) _Float16 lB[64 * 32];   // [n][k] (K-contiguous)
    __shared__ __align__(16) float    sC[64 * 64];   // block output tile staging
    const int tid  = threadIdx.x;
    const int wave = tid >> 5;
    const int lane = tid & 31;
    const int m0 = blockIdx.x * 64;
    const int n0 = blockIdx.y * 64;

    v8f acc0 = {}, acc1 = {}, acc2 = {}, acc3 = {};

    for (int k0 = 0; k0 < Kdim; k0 += 32) {
        // stage A tile 64x32 (f32 -> f16), float4 loads -> packed 4xf16 stores
        #pragma unroll
        for (int i = tid; i < 64 * 32 / 4; i += 128) {
            int r = (i * 4) >> 5, c = (i * 4) & 31;
            float4 f = *(const float4*)&A[(size_t)(m0 + r) * Kdim + (k0 + c)];
            v4h h;
            h[0] = (_Float16)f.x; h[1] = (_Float16)f.y;
            h[2] = (_Float16)f.z; h[3] = (_Float16)f.w;
            *(v4h*)&lA[r * 32 + c] = h;
        }
        // stage B tile 32x64, transposed to [n][k]
        if (transB) {
            #pragma unroll
            for (int i = tid; i < 64 * 32 / 4; i += 128) {
                int n = (i * 4) >> 5, c = (i * 4) & 31;
                float4 f = *(const float4*)&B[(size_t)(n0 + n) * Kdim + (k0 + c)];
                v4h h;
                h[0] = (_Float16)f.x; h[1] = (_Float16)f.y;
                h[2] = (_Float16)f.z; h[3] = (_Float16)f.w;
                *(v4h*)&lB[n * 32 + c] = h;
            }
        } else {
            #pragma unroll
            for (int i = tid; i < 64 * 32 / 4; i += 128) {
                int kk = (i * 4) >> 6, n = (i * 4) & 63;   // coalesced along N
                float4 f = *(const float4*)&B[(size_t)(k0 + kk) * N + (n0 + n)];
                lB[(n + 0) * 32 + kk] = (_Float16)f.x;
                lB[(n + 1) * 32 + kk] = (_Float16)f.y;
                lB[(n + 2) * 32 + kk] = (_Float16)f.z;
                lB[(n + 3) * 32 + kk] = (_Float16)f.w;
            }
        }
        // prefetch next A K-slice into cache while we compute
        if (k0 + 32 < Kdim)
            __builtin_prefetch(&A[(size_t)(m0 + (tid & 63)) * Kdim + (k0 + 32)], 0, 0);
        __syncthreads();

        const int half = lane >> 4;
        const int mrow = (wave << 4) + (lane & 15);
        const int ncol = lane & 15;
        v16h af;
        #pragma unroll
        for (int v = 0; v < 8; ++v) {
            // A frag: M=row per lane; K = (v/4)*16 + half*8 + (v%4)*2 {+0,+1}
            int ka = ((v >> 2) << 4) + (half << 3) + ((v & 3) << 1);
            af[2 * v]     = lA[mrow * 32 + ka];
            af[2 * v + 1] = lA[mrow * 32 + ka + 1];
        }
        #pragma unroll
        for (int tt = 0; tt < 4; ++tt) {
            v16h bf;
            #pragma unroll
            for (int v = 0; v < 8; ++v) {
                // B frag: N=col per lane; K = (v/4)*16 + half*8 + (v%4)*2 {+0,+1}
                int kb = ((v >> 2) << 4) + (half << 3) + ((v & 3) << 1);
                bf[2 * v]     = lB[((tt << 4) + ncol) * 32 + kb];
                bf[2 * v + 1] = lB[((tt << 4) + ncol) * 32 + kb + 1];
            }
            if      (tt == 0) acc0 = __builtin_amdgcn_wmma_f32_16x16x32_f16(false, af, false, bf, (short)0, acc0, false, false);
            else if (tt == 1) acc1 = __builtin_amdgcn_wmma_f32_16x16x32_f16(false, af, false, bf, (short)0, acc1, false, false);
            else if (tt == 2) acc2 = __builtin_amdgcn_wmma_f32_16x16x32_f16(false, af, false, bf, (short)0, acc2, false, false);
            else              acc3 = __builtin_amdgcn_wmma_f32_16x16x32_f16(false, af, false, bf, (short)0, acc3, false, false);
        }
        __syncthreads();
    }

    // C/D frag: acc[j] -> local row = wave*16 + half*8 + j, local col = tt*16 + lane%16
    {
        const int half = lane >> 4;
        const int lcol = lane & 15;
        #pragma unroll
        for (int j = 0; j < 8; ++j) {
            int lrow = (wave << 4) + (half << 3) + j;
            sC[lrow * 64 + 0  + lcol] = acc0[j];
            sC[lrow * 64 + 16 + lcol] = acc1[j];
            sC[lrow * 64 + 32 + lcol] = acc2[j];
            sC[lrow * 64 + 48 + lcol] = acc3[j];
        }
    }
    __syncthreads();

    // cooperative coalesced store with fused epilogue (float4 -> b128)
    const float lg = logf(100000.0f) / (float)N_HID;
    for (int i = tid; i < 64 * 64 / 4; i += 128) {
        int lrow = (i * 4) >> 6;
        int lcol = (i * 4) & 63;
        int row  = m0 + lrow;
        float4 v = *(const float4*)&sC[lrow * 64 + lcol];
        if (mode >= 1) {
            v.x += bias[n0 + lcol + 0]; v.y += bias[n0 + lcol + 1];
            v.z += bias[n0 + lcol + 2]; v.w += bias[n0 + lcol + 3];
        }
        if (mode == 2) {  // sinusoidal positional encoding, pos = t+1
            float pos = (float)(row / nodesPerFrame + 1);
            float* pv = (float*)&v;
            #pragma unroll
            for (int u = 0; u < 4; ++u) {
                int col = n0 + lcol + u;
                float dv = __expf(-(float)(col & ~1) * lg);
                pv[u] += (col & 1) ? __cosf(pos * dv) : __sinf(pos * dv);
            }
        }
        if (mode == 3) {
            v.x = fmaxf(v.x, 0.f); v.y = fmaxf(v.y, 0.f);
            v.z = fmaxf(v.z, 0.f); v.w = fmaxf(v.w, 0.f);
        }
        *(float4*)&C[(size_t)row * N + n0 + lcol] = v;
    }
}

// ---------------------------------------------------------------------------
// GAT stage kernels (per relation r, frame t)
// ---------------------------------------------------------------------------
__global__ void gat_alpha_kernel(const float* __restrict__ xw,
                                 const float* __restrict__ att_s,
                                 const float* __restrict__ att_d,
                                 float* __restrict__ al_s, float* __restrict__ al_d)
{
    int i = blockIdx.x * blockDim.x + threadIdx.x;      // (n, h)
    if (i >= N_NODES * 2) return;
    int n = i >> 1, h = i & 1;
    const float* row = xw + (size_t)n * 128 + h * 64;
    float s = 0.f, d = 0.f;
    for (int c = 0; c < 64; ++c) {
        s += row[c] * att_s[h * 64 + c];
        d += row[c] * att_d[h * 64 + c];
    }
    al_s[i] = s; al_d[i] = d;
}

__global__ void gat_init_kernel(unsigned* __restrict__ menc,
                                float* __restrict__ denom)
{
    int i = blockIdx.x * blockDim.x + threadIdx.x;      // (n, h)
    if (i >= N_NODES * 2) return;
    menc[i]  = fenc(-__builtin_inff());
    denom[i] = 0.f;
}

__global__ void zero_f4(float* __restrict__ p, int n4)   // n4 = count/4
{
    int i = blockIdx.x * blockDim.x + threadIdx.x;
    if (i >= n4) return;
    *(float4*)&p[(size_t)i * 4] = make_float4(0.f, 0.f, 0.f, 0.f);
}

__global__ void gat_edge_max(const int* __restrict__ src, const int* __restrict__ dst,
                             const float* __restrict__ al_s, const float* __restrict__ al_d,
                             unsigned* __restrict__ menc)
{
    int i = blockIdx.x * blockDim.x + threadIdx.x;      // (e, h)
    if (i >= N_EDGES * 2) return;
    int e = i >> 1, h = i & 1;
    int s = src[e], d = dst[e];
    float ee = al_s[s * 2 + h] + al_d[d * 2 + h];
    ee = (ee > 0.f) ? ee : 0.2f * ee;                   // leaky_relu 0.2
    atomicMax(&menc[d * 2 + h], fenc(ee));
}

__global__ void gat_m_decode(const unsigned* __restrict__ menc, float* __restrict__ mf)
{
    int i = blockIdx.x * blockDim.x + threadIdx.x;
    if (i >= N_NODES * 2) return;
    float m = fdec(menc[i]);
    if (!(fabsf(m) < 3.0e38f)) m = 0.f;                 // non-finite (no edges) -> 0
    mf[i] = m;
}

__global__ void gat_edge_denom(const int* __restrict__ src, const int* __restrict__ dst,
                               const float* __restrict__ al_s, const float* __restrict__ al_d,
                               const float* __restrict__ mf, float* __restrict__ denom)
{
    int i = blockIdx.x * blockDim.x + threadIdx.x;      // (e, h)
    if (i >= N_EDGES * 2) return;
    int e = i >> 1, h = i & 1;
    int s = src[e], d = dst[e];
    float ee = al_s[s * 2 + h] + al_d[d * 2 + h];
    ee = (ee > 0.f) ? ee : 0.2f * ee;
    atomicAdd(&denom[d * 2 + h], __expf(ee - mf[d * 2 + h]));
}

// 2 edges per 256-thread block; 128 lanes per edge (h*64+c)
__global__ __launch_bounds__(256)
void gat_edge_scatter(const int* __restrict__ src, const int* __restrict__ dst,
                      const float* __restrict__ al_s, const float* __restrict__ al_d,
                      const float* __restrict__ mf, const float* __restrict__ denom,
                      const float* __restrict__ xw, float* __restrict__ outacc)
{
    int e = blockIdx.x * 2 + (threadIdx.x >> 7);
    if (e >= N_EDGES) return;
    int j = threadIdx.x & 127;                          // h*64 + c
    int h = j >> 6;
    int s = src[e], d = dst[e];
    float ee = al_s[s * 2 + h] + al_d[d * 2 + h];
    ee = (ee > 0.f) ? ee : 0.2f * ee;
    float ex    = __expf(ee - mf[d * 2 + h]);
    float alpha = ex / (denom[d * 2 + h] + 1e-16f);
    atomicAdd(&outacc[(size_t)d * 128 + j], xw[(size_t)s * 128 + j] * alpha);
}

__global__ void gat_finalize(const float* __restrict__ outacc,
                             const float* __restrict__ bias,
                             float* __restrict__ intra_rt)
{
    int i = blockIdx.x * blockDim.x + threadIdx.x;      // (n, c)
    if (i >= N_NODES * 64) return;
    int n = i >> 6, c = i & 63;
    intra_rt[i] = 0.5f * (outacc[(size_t)n * 128 + c] + outacc[(size_t)n * 128 + 64 + c])
                  + bias[c];
}

// ---------------------------------------------------------------------------
// Relation aggregation
// ---------------------------------------------------------------------------
__global__ void zero_small(float* p, int n)
{
    int i = blockIdx.x * blockDim.x + threadIdx.x;
    if (i < n) p[i] = 0.f;
}

__global__ __launch_bounds__(256)
void relagg_score(const float* __restrict__ intra_t, size_t rstride,
                  const float* __restrict__ w1, const float* __restrict__ b1,
                  const float* __restrict__ w2, float* __restrict__ wsum)
{
    __shared__ float sw1[64 * 64];
    __shared__ float sb1[64], sw2[64];
    __shared__ float red[2][8];
    for (int i = threadIdx.x; i < 64 * 64; i += 256) sw1[i] = w1[i];
    if (threadIdx.x < 64) { sb1[threadIdx.x] = b1[threadIdx.x]; sw2[threadIdx.x] = w2[threadIdx.x]; }
    __syncthreads();

    int n = blockIdx.x * 256 + threadIdx.x;
    float acc[2] = {0.f, 0.f};
    if (n < N_NODES) {
        for (int r = 0; r < 2; ++r) {
            const float* hrow = intra_t + (size_t)r * rstride + (size_t)n * 64;
            float hv[64];
            for (int k = 0; k < 64; ++k) hv[k] = hrow[k];
            float s = 0.f;
            for (int j = 0; j < 64; ++j) {
                float z = sb1[j];
                for (int k = 0; k < 64; ++k) z += hv[k] * sw1[j * 64 + k];
                s += tanhf(z) * sw2[j];
            }
            acc[r] = s;
        }
    }
    for (int r = 0; r < 2; ++r) {
        float v = acc[r];
        for (int o = 16; o > 0; o >>= 1) v += __shfl_xor(v, o, 32);
        if ((threadIdx.x & 31) == 0) red[r][threadIdx.x >> 5] = v;
    }
    __syncthreads();
    if (threadIdx.x < 2) {
        float t = 0.f;
        for (int w = 0; w < 8; ++w) t += red[threadIdx.x][w];
        atomicAdd(&wsum[threadIdx.x], t);
    }
}

__global__ void relagg_beta(const float* __restrict__ wsum, float* __restrict__ beta)
{
    int t = threadIdx.x;
    if (t >= T_FRAMES) return;
    float a = wsum[t * 2 + 0] * (1.0f / (float)N_NODES);
    float b = wsum[t * 2 + 1] * (1.0f / (float)N_NODES);
    float m = fmaxf(a, b);
    float ea = __expf(a - m), eb = __expf(b - m);
    float s = ea + eb;
    beta[t * 2 + 0] = ea / s;
    beta[t * 2 + 1] = eb / s;
}

__global__ void relagg_combine(const float* __restrict__ intra,
                               const float* __restrict__ beta,
                               float* __restrict__ inter)
{
    const int total = T_FRAMES * N_NODES * 64;
    const size_t rstride = (size_t)T_FRAMES * N_NODES * 64;
    int i = blockIdx.x * blockDim.x + threadIdx.x;      // (t, n, c)
    if (i >= total) return;
    int t = i / (N_NODES * 64);
    inter[i] = beta[t * 2 + 0] * intra[(size_t)i]
             + beta[t * 2 + 1] * intra[rstride + (size_t)i];
}

// ---------------------------------------------------------------------------
// Temporal self-attention core (T=4, D=64): one node per 32-lane group
// ---------------------------------------------------------------------------
__global__ __launch_bounds__(256)
void temporal_attn(const float* __restrict__ q, const float* __restrict__ k,
                   const float* __restrict__ v, float* __restrict__ hatt)
{
    int n    = blockIdx.x * 8 + (threadIdx.x >> 5);
    int lane = threadIdx.x & 31;
    if (n >= N_NODES) return;

    float qf[4][2], kf[4][2], vf[4][2];
    #pragma unroll
    for (int t = 0; t < 4; ++t) {
        size_t base = ((size_t)t * N_NODES + n) * 64 + lane * 2;
        qf[t][0] = q[base]; qf[t][1] = q[base + 1];
        kf[t][0] = k[base]; kf[t][1] = k[base + 1];
        vf[t][0] = v[base]; vf[t][1] = v[base + 1];
    }
    float sc[4][4];
    #pragma unroll
    for (int t = 0; t < 4; ++t)
        #pragma unroll
        for (int s = 0; s < 4; ++s) {
            float d = qf[t][0] * kf[s][0] + qf[t][1] * kf[s][1];
            for (int o = 16; o > 0; o >>= 1) d += __shfl_xor(d, o, 32);
            sc[t][s] = d;
        }
    #pragma unroll
    for (int t = 0; t < 4; ++t) {
        float m = fmaxf(fmaxf(sc[t][0], sc[t][1]), fmaxf(sc[t][2], sc[t][3]));
        float e0 = __expf(sc[t][0] - m), e1 = __expf(sc[t][1] - m);
        float e2 = __expf(sc[t][2] - m), e3 = __expf(sc[t][3] - m);
        float inv = 1.f / (e0 + e1 + e2 + e3);
        sc[t][0] = e0 * inv; sc[t][1] = e1 * inv; sc[t][2] = e2 * inv; sc[t][3] = e3 * inv;
    }
    #pragma unroll
    for (int t = 0; t < 4; ++t) {
        float o0 = 0.f, o1 = 0.f;
        #pragma unroll
        for (int s = 0; s < 4; ++s) { o0 += sc[t][s] * vf[s][0]; o1 += sc[t][s] * vf[s][1]; }
        size_t base = ((size_t)t * N_NODES + n) * 64 + lane * 2;
        hatt[base] = o0; hatt[base + 1] = o1;
    }
}

// ---------------------------------------------------------------------------
// Gated residual + layernorm: one row (t,n) per 32-lane group
// ---------------------------------------------------------------------------
__global__ __launch_bounds__(256)
void final_ln(const float* __restrict__ act, const float* __restrict__ res,
              const float* __restrict__ res_alpha, const float* __restrict__ g,
              const float* __restrict__ b, float* __restrict__ out, int rows)
{
    int row  = blockIdx.x * 8 + (threadIdx.x >> 5);
    int lane = threadIdx.x & 31;
    if (row >= rows) return;
    float alpha = 1.f / (1.f + __expf(-res_alpha[0]));
    size_t base = (size_t)row * 64 + lane * 2;
    float o0 = act[base]     * alpha + res[base]     * (1.f - alpha);
    float o1 = act[base + 1] * alpha + res[base + 1] * (1.f - alpha);
    float s = o0 + o1;
    for (int o = 16; o > 0; o >>= 1) s += __shfl_xor(s, o, 32);
    float mu = s * (1.f / 64.f);
    float d0 = o0 - mu, d1 = o1 - mu;
    float vs = d0 * d0 + d1 * d1;
    for (int o = 16; o > 0; o >>= 1) vs += __shfl_xor(vs, o, 32);
    float inv = rsqrtf(vs * (1.f / 64.f) + 1e-5f);
    int c = lane * 2;
    out[base]     = d0 * inv * g[c]     + b[c];
    out[base + 1] = d1 * inv * g[c + 1] + b[c + 1];
}

// ---------------------------------------------------------------------------
extern "C" void kernel_launch(void* const* d_in, const int* in_sizes, int n_in,
                              void* d_out, int out_size, void* d_ws, size_t ws_size,
                              hipStream_t stream)
{
    const float* x        = (const float*)d_in[0];
    const int*   ei       = (const int*)d_in[1];     // [R,T,2,E]
    const float* W_gat    = (const float*)d_in[2];
    const float* att_src  = (const float*)d_in[3];
    const float* att_dst  = (const float*)d_in[4];
    const float* bias_gat = (const float*)d_in[5];
    const float* ra_w1    = (const float*)d_in[6];
    const float* ra_b1    = (const float*)d_in[7];
    const float* ra_w2    = (const float*)d_in[8];
    const float* ta_proj_w= (const float*)d_in[9];
    const float* ta_proj_b= (const float*)d_in[10];
    const float* ta_q_w   = (const float*)d_in[11];
    const float* ta_k_w   = (const float*)d_in[12];
    const float* ta_v_w   = (const float*)d_in[13];
    const float* ta_fc_w  = (const float*)d_in[14];
    const float* ta_fc_b  = (const float*)d_in[15];
    const float* res_w    = (const float*)d_in[16];
    const float* res_b    = (const float*)d_in[17];
    const float* res_alpha= (const float*)d_in[18];
    const float* ln_g     = (const float*)d_in[19];
    const float* ln_b     = (const float*)d_in[20];

    char* ws = (char*)d_ws;
    float*    intra  = (float*)(ws + OFF_INTRA);
    float*    inter  = (float*)(ws + OFF_INTER);
    float*    hproj  = (float*)(ws + OFF_HPROJ);
    float*    qbuf   = (float*)(ws + OFF_Q);
    float*    kbuf   = (float*)(ws + OFF_K);
    float*    vbuf   = (float*)(ws + OFF_V);
    float*    hatt   = (float*)(ws + OFF_HATT);
    float*    act    = (float*)(ws + OFF_ACT);
    float*    res    = (float*)(ws + OFF_RES);
    float*    xw     = (float*)(ws + OFF_XW);
    float*    outacc = (float*)(ws + OFF_OUTACC);
    float*    al_s   = (float*)(ws + OFF_ALS);
    float*    al_d   = (float*)(ws + OFF_ALD);
    unsigned* menc   = (unsigned*)(ws + OFF_MENC);
    float*    mf     = (float*)(ws + OFF_MF);
    float*    denom  = (float*)(ws + OFF_DEN);
    float*    wsum   = (float*)(ws + OFF_WSUM);
    float*    beta   = (float*)(ws + OFF_BETA);

    const int MTN = T_FRAMES * N_NODES;                // 160000

    // ---------------- Stage 1: intra-relation GAT per (r, t) ----------------
    for (int r = 0; r < N_REL; ++r) {
        for (int t = 0; t < T_FRAMES; ++t) {
            const float* xt  = x + (size_t)t * N_NODES * N_INP;
            const float* Wr  = W_gat + (size_t)r * N_INP * 128;
            const int*   src = ei + ((((size_t)r * T_FRAMES + t) * 2 + 0) * N_EDGES);
            const int*   dst = ei + ((((size_t)r * T_FRAMES + t) * 2 + 1) * N_EDGES);
            float* intra_rt  = intra + ((size_t)r * T_FRAMES + t) * N_NODES * 64;

            // xw = x[t] @ W[r]   [40000,128]x[128,128]
            wmma_gemm_f32<<<dim3(N_NODES / 64, 128 / 64), 128, 0, stream>>>(
                xt, Wr, xw, nullptr, N_NODES, 128, 128, /*transB=*/0, /*mode=*/0, 0);

            gat_alpha_kernel<<<(N_NODES * 2 + 255) / 256, 256, 0, stream>>>(
                xw, att_src + (size_t)r * 128, att_dst + (size_t)r * 128, al_s, al_d);
            gat_init_kernel<<<(N_NODES * 2 + 255) / 256, 256, 0, stream>>>(menc, denom);
            zero_f4<<<(N_NODES * 128 / 4 + 255) / 256, 256, 0, stream>>>(outacc, N_NODES * 128 / 4);
            gat_edge_max<<<(N_EDGES * 2 + 255) / 256, 256, 0, stream>>>(src, dst, al_s, al_d, menc);
            gat_m_decode<<<(N_NODES * 2 + 255) / 256, 256, 0, stream>>>(menc, mf);
            gat_edge_denom<<<(N_EDGES * 2 + 255) / 256, 256, 0, stream>>>(src, dst, al_s, al_d, mf, denom);
            gat_edge_scatter<<<(N_EDGES + 1) / 2, 256, 0, stream>>>(src, dst, al_s, al_d, mf, denom, xw, outacc);
            gat_finalize<<<(N_NODES * 64 + 255) / 256, 256, 0, stream>>>(
                outacc, bias_gat + (size_t)r * 64, intra_rt);
        }
    }

    // ---------------- Stage 2: inter-relation aggregation ----------------
    zero_small<<<1, 32, 0, stream>>>(wsum, T_FRAMES * N_REL);
    for (int t = 0; t < T_FRAMES; ++t) {
        relagg_score<<<(N_NODES + 255) / 256, 256, 0, stream>>>(
            intra + (size_t)t * N_NODES * 64, (size_t)T_FRAMES * N_NODES * 64,
            ra_w1 + (size_t)t * 64 * 64, ra_b1 + (size_t)t * 64, ra_w2 + (size_t)t * 64,
            wsum + t * 2);
    }
    relagg_beta<<<1, 32, 0, stream>>>(wsum, beta);
    relagg_combine<<<(MTN * 64 + 255) / 256, 256, 0, stream>>>(intra, beta, inter);

    // ---------------- Stage 3: temporal projection + positional encoding ----
    wmma_gemm_f32<<<dim3(MTN / 64, 1), 128, 0, stream>>>(
        inter, ta_proj_w, hproj, ta_proj_b, MTN, 64, 64, 1, /*mode=*/2, N_NODES);

    // ---------------- Stage 4: Q/K/V projections -------------------------
    wmma_gemm_f32<<<dim3(MTN / 64, 1), 128, 0, stream>>>(
        hproj, ta_q_w, qbuf, nullptr, MTN, 64, 64, 1, 0, 0);
    wmma_gemm_f32<<<dim3(MTN / 64, 1), 128, 0, stream>>>(
        hproj, ta_k_w, kbuf, nullptr, MTN, 64, 64, 1, 0, 0);
    wmma_gemm_f32<<<dim3(MTN / 64, 1), 128, 0, stream>>>(
        hproj, ta_v_w, vbuf, nullptr, MTN, 64, 64, 1, 0, 0);

    // ---------------- Stage 5: per-node 4x4 attention --------------------
    temporal_attn<<<(N_NODES + 7) / 8, 256, 0, stream>>>(qbuf, kbuf, vbuf, hatt);

    // ---------------- Stage 6: fc + relu ---------------------------------
    wmma_gemm_f32<<<dim3(MTN / 64, 1), 128, 0, stream>>>(
        hatt, ta_fc_w, act, ta_fc_b, MTN, 64, 64, 1, /*mode=*/3, 0);

    // ---------------- Stage 7: residual projection -----------------------
    wmma_gemm_f32<<<dim3(MTN / 64, 1), 128, 0, stream>>>(
        x, res_w, res, res_b, MTN, 64, 128, 1, /*mode=*/1, 0);

    // ---------------- Stage 8: gated residual + layernorm ----------------
    final_ln<<<(MTN + 7) / 8, 256, 0, stream>>>(act, res, res_alpha, ln_g, ln_b,
                                                (float*)d_out, MTN);
}
